// TLSTM_13185549598728
// MI455X (gfx1250) — compile-verified, matrix operand order checked by reading, not run on hardware
//
#include <hip/hip_runtime.h>
#include <hip/hip_bf16.h>
#include <math.h>

typedef __attribute__((ext_vector_type(16))) _Float16 v16h;
typedef __attribute__((ext_vector_type(8)))  float    v8f;
typedef __attribute__((ext_vector_type(4)))  int      v4i;
typedef _Float16 half_t;

constexpr int kB   = 256;
constexpr int kSeq = 200;
constexpr int kD   = 256;
constexpr int kDG  = 1024;   // 4*D gate width
constexpr int kK2  = 512;    // 2*D concat [h,x]
constexpr size_t kBSD = (size_t)kB * kSeq * kD;   // 13,107,200 elements

// ---------------------------------------------------------------------------
// gfx1250 async global->LDS copy (guarded: falls back to plain 16B copy)
// builtin signature on this toolchain: (v4i AS1*, v4i AS3*, imm offset, imm cpol)
// ---------------------------------------------------------------------------
#if defined(__gfx1250__) && __has_builtin(__builtin_amdgcn_global_load_async_to_lds_b128)
#define HAVE_ASYNC_LDS 1
typedef __attribute__((address_space(1))) v4i gv4i_t;
typedef __attribute__((address_space(3))) v4i lv4i_t;
#endif

__device__ __forceinline__ void async_copy16(const void* g, void* l) {
#ifdef HAVE_ASYNC_LDS
  __builtin_amdgcn_global_load_async_to_lds_b128((gv4i_t*)g, (lv4i_t*)l, 0, 0);
#else
  *(uint4*)l = *(const uint4*)g;
#endif
}
__device__ __forceinline__ void async_join() {
#ifdef HAVE_ASYNC_LDS
#if __has_builtin(__builtin_amdgcn_s_wait_asynccnt)
  __builtin_amdgcn_s_wait_asynccnt(0);
#else
  asm volatile("s_wait_asynccnt 0" ::: "memory");
#endif
#endif
}

__device__ __forceinline__ float sigmoidf_(float x) {
  return 1.0f / (1.0f + __expf(-x));
}

// ---------------------------------------------------------------------------
// Conversions / weight packing
// ---------------------------------------------------------------------------
__global__ __launch_bounds__(256) void k_f32_to_f16(const float* __restrict__ a,
                                                    half_t* __restrict__ o) {
  size_t i = (size_t)blockIdx.x * 256 + threadIdx.x;
  o[i] = (half_t)a[i];
}

// WT[n][k] = (f16) W[k][n]    (B-operand: per-lane contiguous along K)
__global__ __launch_bounds__(256) void k_transposeW(const float* __restrict__ W,
                                                    half_t* __restrict__ WT,
                                                    int K, int N) {
  int n = blockIdx.x;
  for (int k = threadIdx.x; k < K; k += blockDim.x)
    WT[(size_t)n * K + k] = (half_t)W[(size_t)k * N + n];
}

// WcatT[n][k] : rows k<256 from Wall_w, k>=256 from Uall_w  (N=1024, K=512)
__global__ __launch_bounds__(256) void k_buildWcat(const float* __restrict__ Wall,
                                                   const float* __restrict__ Uall,
                                                   half_t* __restrict__ WT) {
  int n = blockIdx.x;
  for (int k = threadIdx.x; k < kK2; k += blockDim.x) {
    float v = (k < kD) ? Wall[(size_t)k * kDG + n] : Uall[(size_t)(k - kD) * kDG + n];
    WT[(size_t)n * kK2 + k] = (half_t)v;
  }
}

__global__ __launch_bounds__(256) void k_biascat(const float* __restrict__ a,
                                                 const float* __restrict__ b,
                                                 float* __restrict__ o) {
  int i = blockIdx.x * 256 + threadIdx.x;
  o[i] = a[i] + b[i];
}

// ---------------------------------------------------------------------------
// Generic WMMA GEMM: C[M][N] = A[M][K] * B[K][N] + bias   (BT stored [N][K] f16)
// one wave per 16x16 output tile
// ---------------------------------------------------------------------------
__global__ __launch_bounds__(32)
void k_gemm16(const half_t* __restrict__ A, const half_t* __restrict__ BT,
              const float* __restrict__ bias, float* __restrict__ C,
              half_t* __restrict__ C16, int N, int K) {
  const int mt = blockIdx.x, nt = blockIdx.y;
  const int lane = threadIdx.x;
  const int mr = lane & 15, kh = lane >> 4;
  v8f acc = {};
  const half_t* ap = A  + ((size_t)(mt * 16 + mr)) * K + kh * 16;
  const half_t* bp = BT + ((size_t)(nt * 16 + mr)) * K + kh * 16;
  for (int k = 0; k < K; k += 32) {
    __builtin_prefetch(bp + k + 128, 0, 0);
    v16h a = *(const v16h*)(ap + k);
    v16h b = *(const v16h*)(bp + k);
    acc = __builtin_amdgcn_wmma_f32_16x16x32_f16(false, a, false, b, (short)0,
                                                 acc, false, false);
  }
  const int n  = nt * 16 + (lane & 15);
  const int mb = mt * 16 + (lane >> 4) * 8;
  const float bv = bias[n];
#pragma unroll
  for (int r = 0; r < 8; ++r) {
    float v = acc[r] + bv;
    size_t idx = (size_t)(mb + r) * N + n;
    C[idx] = v;
    if (C16) C16[idx] = (half_t)v;
  }
}

// ---------------------------------------------------------------------------
// time_feature = (1 - tanh((t/180 * sel_w + sel_b)^2)) @ time_w + time_b
// ---------------------------------------------------------------------------
__global__ __launch_bounds__(256)
void k_timefeat(const float* __restrict__ seq_time, const float* __restrict__ sel_w,
                const float* __restrict__ sel_b, const float* __restrict__ time_w,
                const float* __restrict__ time_b, float* __restrict__ tf) {
  const int bs = blockIdx.x;                       // b*SEQ + s
  __shared__ float tb[64];
  const float t = seq_time[bs] * (1.0f / 180.0f);
  if (threadIdx.x < 64) {
    float z = t * sel_w[threadIdx.x] + sel_b[threadIdx.x];
    tb[threadIdx.x] = 1.0f - tanhf(z * z);
  }
  __syncthreads();
  const int n = threadIdx.x;
  float acc = time_b[n];
#pragma unroll 8
  for (int j = 0; j < 64; ++j) acc += tb[j] * time_w[j * kD + n];
  tf[(size_t)bs * kD + n] = acc;
}

// ---------------------------------------------------------------------------
// Persistent T-LSTM scan. grid = 16 blocks (16 batch rows each), 512 thr = 16 waves.
// Wave w owns gate columns [w*16, w*16+16) of all four gates -> cell update in regs.
// ---------------------------------------------------------------------------
__global__ __launch_bounds__(512)
void k_tlstm(const half_t* __restrict__ x16,      // [B][SEQ][D] f16
             const float*  __restrict__ tf,       // [B][SEQ][D] f32
             const half_t* __restrict__ WcatT,    // [1024][512] f16
             const float*  __restrict__ biascat,  // [1024]
             const half_t* __restrict__ WdT,      // [256][256] f16
             const float*  __restrict__ Wd_b,     // [256]
             float* __restrict__ outs,            // [B][SEQ][D] f32
             half_t* __restrict__ outs16)         // optional f16 copy
{
  __shared__ half_t hx[16][kK2];    // [m][k] : k<256 = h_{t-1}, k>=256 = x_t
  __shared__ half_t c16s[16][kD];   // c_{t-1} as f16 (A operand for c@Wd)
  __shared__ float  tsb[16][kD];    // time feature tile

  const int tid  = threadIdx.x;
  const int lane = tid & 31;
  const int wave = tid >> 5;        // 0..15
  const int b0   = blockIdx.x * 16;
  const int n0   = wave * 16;
  const int mrow = lane & 15;       // A row / B col within tile
  const int kh   = lane >> 4;       // K half (0/1)
  const int nEl  = n0 + (lane & 15);
  const int mEl  = (lane >> 4) * 8;

  for (int i = tid; i < 16 * kK2; i += 512) (&hx[0][0])[i] = (half_t)0.0f;
  for (int i = tid; i < 16 * kD;  i += 512) (&c16s[0][0])[i] = (half_t)0.0f;
  float c_r[8];
#pragma unroll
  for (int r = 0; r < 8; ++r) c_r[r] = 0.0f;
  __syncthreads();

  const float bdW = Wd_b[nEl];
  const float bf  = biascat[nEl];
  const float bi  = biascat[kD + nEl];
  const float bo  = biascat[2 * kD + nEl];
  const float bc  = biascat[3 * kD + nEl];

  for (int t = 0; t < kSeq; ++t) {
    // -- stage x_t (f16, 8KB) and ts_t (f32, 16KB) into LDS (async path)
    {
      const int m = tid >> 5;                 // 32 threads per row
      const int c = tid & 31;
      const half_t* gx = x16 + ((size_t)(b0 + m) * kSeq + t) * kD + c * 8;
      async_copy16(gx, &hx[m][kD + c * 8]);
      const float* gt = tf + ((size_t)(b0 + m) * kSeq + t) * kD + c * 8;
      async_copy16(gt,     &tsb[m][c * 8]);
      async_copy16(gt + 4, &tsb[m][c * 8 + 4]);
    }
    async_join();
    __syncthreads();

    // -- c_s1 = tanh(c @ Wd + b);  c_adj = c - c_s1 + c_s1*ts   (in registers)
    v8f accd = {};
#pragma unroll
    for (int kk = 0; kk < kD / 32; ++kk) {
      v16h a  = *(const v16h*)&c16s[mrow][kk * 32 + kh * 16];
      v16h bm = *(const v16h*)&WdT[(size_t)nEl * kD + kk * 32 + kh * 16];
      accd = __builtin_amdgcn_wmma_f32_16x16x32_f16(false, a, false, bm, (short)0,
                                                    accd, false, false);
    }
    float cadj[8];
#pragma unroll
    for (int r = 0; r < 8; ++r) {
      float cs1 = tanhf(accd[r] + bdW);
      cadj[r] = c_r[r] - cs1 + cs1 * tsb[mEl + r][nEl];
    }

    // -- gates = sigmoid([h,x] @ [Wall;Uall] + bias); 4 gate tiles per wave
    v8f accg[4] = {};
    for (int kk = 0; kk < kK2 / 32; ++kk) {
      v16h a = *(const v16h*)&hx[mrow][kk * 32 + kh * 16];
#pragma unroll
      for (int g = 0; g < 4; ++g) {
        const half_t* bp = WcatT + (size_t)(g * kD + nEl) * kK2 + kk * 32 + kh * 16;
        __builtin_prefetch(bp + 32, 0, 0);
        v16h bm = *(const v16h*)bp;
        accg[g] = __builtin_amdgcn_wmma_f32_16x16x32_f16(false, a, false, bm, (short)0,
                                                         accg[g], false, false);
      }
    }

    // -- LSTM cell update, fully in registers
    float hval[8];
#pragma unroll
    for (int r = 0; r < 8; ++r) {
      float f  = sigmoidf_(accg[0][r] + bf);
      float ii = sigmoidf_(accg[1][r] + bi);
      float o  = sigmoidf_(accg[2][r] + bo);
      float ct = sigmoidf_(accg[3][r] + bc);
      float cn = f * cadj[r] + ii * ct;
      c_r[r]  = cn;
      hval[r] = o * tanhf(cn);
    }
    __syncthreads();   // all LDS reads of this step done before rewrites
#pragma unroll
    for (int r = 0; r < 8; ++r) {
      const int m = mEl + r;
      c16s[m][nEl] = (half_t)c_r[r];
      hx[m][nEl]   = (half_t)hval[r];
      size_t gi = ((size_t)(b0 + m) * kSeq + t) * kD + nEl;
      outs[gi] = hval[r];
      if (outs16) outs16[gi] = (half_t)hval[r];
    }
    __syncthreads();   // publish h/c before next step's GEMMs
  }
}

// ---------------------------------------------------------------------------
// aligned[b][0][:] = x[b][0][:]
// ---------------------------------------------------------------------------
__global__ __launch_bounds__(256)
void k_copy_ek(const float* __restrict__ x, float* __restrict__ aligned) {
  const int b = blockIdx.x;
  aligned[(size_t)b * kSeq * kD + threadIdx.x] = x[(size_t)b * kSeq * kD + threadIdx.x];
}

// ---------------------------------------------------------------------------
// attention + aligned rows 1..199 : one block (64 thr) per (b,s), s in [0,199)
// ---------------------------------------------------------------------------
__global__ __launch_bounds__(64)
void k_attn(const float* __restrict__ x, const float* __restrict__ whk,
            const float* __restrict__ w1_w, const float* __restrict__ w1_b,
            const float* __restrict__ w2_w, const float* __restrict__ w2_b,
            float* __restrict__ aligned) {
  const int s = blockIdx.x;          // 0..198
  const int b = blockIdx.y;
  __shared__ float vbuf[64];
  __shared__ float lg[2];
  const int j = threadIdx.x;
  const float* ek = x   + (size_t)b * kSeq * kD;               // x[b,0,:]
  const float* wr = whk + ((size_t)b * kSeq + s) * kD;         // whk[b,s,:]
  float acc = w1_b[j];
  for (int k = 0; k < kD; ++k) acc += ek[k] * w1_w[(size_t)k * 64 + j];
  for (int k = 0; k < kD; ++k) acc += wr[k] * w1_w[(size_t)(kD + k) * 64 + j];
  vbuf[j] = tanhf(acc);
  __syncthreads();
  if (j < 2) {
    float l = w2_b[j];
    for (int k = 0; k < 64; ++k) l += vbuf[k] * w2_w[k * 2 + j];
    lg[j] = l;
  }
  __syncthreads();
  const float m  = fmaxf(lg[0], lg[1]);
  const float e0 = __expf(lg[0] - m), e1 = __expf(lg[1] - m);
  const float a0 = e0 / (e0 + e1),    a1 = e1 / (e0 + e1);
  float* ar = aligned + ((size_t)b * kSeq + (s + 1)) * kD;
  for (int k = j; k < kD; k += 64) ar[k] = ek[k] * a0 + wr[k] * a1;
}

// gen_x = aligned + (normal_noise - predicted_noise), stored f16
__global__ __launch_bounds__(256)
void k_genx(const float* __restrict__ aligned, const float* __restrict__ nn,
            const float* __restrict__ pn, half_t* __restrict__ g16) {
  size_t i = (size_t)blockIdx.x * 256 + threadIdx.x;
  g16[i] = (half_t)(aligned[i] + (nn[i] - pn[i]));
}

// out[b,:] = max_s(hs[b,s,:]) @ out_w + out_b
__global__ __launch_bounds__(256)
void k_head(const float* __restrict__ hs, const float* __restrict__ out_w,
            const float* __restrict__ out_b, float* __restrict__ o) {
  const int b = blockIdx.x, d = threadIdx.x;
  __shared__ float md[kD];
  float mx = -1e30f;
  for (int s = 0; s < kSeq; ++s) mx = fmaxf(mx, hs[((size_t)b * kSeq + s) * kD + d]);
  md[d] = mx;
  __syncthreads();
  if (d < 2) {
    float acc = out_b[d];
    for (int k = 0; k < kD; ++k) acc += md[k] * out_w[k * 2 + d];
    o[b * 2 + d] = acc;
  }
}

// ---------------------------------------------------------------------------
extern "C" void kernel_launch(void* const* d_in, const int* in_sizes, int n_in,
                              void* d_out, int out_size, void* d_ws, size_t ws_size,
                              hipStream_t stream) {
  (void)in_sizes; (void)n_in; (void)out_size; (void)ws_size;
  const float* input_seqs = (const float*)d_in[0];
  const float* seq_time   = (const float*)d_in[3];
  const float* normal_n   = (const float*)d_in[6];
  const float* pred_n     = (const float*)d_in[7];
  const float* emb2_w = (const float*)d_in[9],  *emb2_b = (const float*)d_in[10];
  const float* Wall_w = (const float*)d_in[11], *Wall_b = (const float*)d_in[12];
  const float* Uall_w = (const float*)d_in[13], *Uall_b = (const float*)d_in[14];
  const float* Wd_w   = (const float*)d_in[15], *Wd_b   = (const float*)d_in[16];
  const float* time_w = (const float*)d_in[17], *time_b = (const float*)d_in[18];
  const float* sel_w  = (const float*)d_in[19], *sel_b  = (const float*)d_in[20];
  const float* whk_w  = (const float*)d_in[21], *whk_b  = (const float*)d_in[22];
  const float* w1_w   = (const float*)d_in[23], *w1_b   = (const float*)d_in[24];
  const float* w2_w   = (const float*)d_in[25], *w2_b   = (const float*)d_in[26];
  const float* out_w  = (const float*)d_in[27], *out_b  = (const float*)d_in[28];

  char* ws = (char*)d_ws;
  size_t off = 0;
  auto alloc = [&](size_t bytes) -> char* {
    char* p = ws + off;
    off += (bytes + 255) & ~(size_t)255;
    return p;
  };
  half_t* inp16   = (half_t*)alloc(kBSD * 2);
  half_t* x16     = (half_t*)alloc(kBSD * 2);
  float*  x32     = (float*) alloc(kBSD * 4);
  float*  tf32    = (float*) alloc(kBSD * 4);
  float*  outs32  = (float*) alloc(kBSD * 4);
  half_t* outs16  = (half_t*)alloc(kBSD * 2);
  float*  whk32   = (float*) alloc(kBSD * 4);
  float*  align32 = (float*) alloc(kBSD * 4);
  half_t* genx16  = (half_t*)alloc(kBSD * 2);
  float*  gouts32 = (float*) alloc(kBSD * 4);
  half_t* emb2T   = (half_t*)alloc((size_t)kD * kD * 2);
  half_t* WdT     = (half_t*)alloc((size_t)kD * kD * 2);
  half_t* whkT    = (half_t*)alloc((size_t)kD * kD * 2);
  half_t* WcatT   = (half_t*)alloc((size_t)kDG * kK2 * 2);
  float*  biascat = (float*) alloc((size_t)kDG * 4);

  const int rows = kB * kSeq;                         // 51200
  // 1. pack inputs/weights into f16
  k_f32_to_f16<<<(int)(kBSD / 256), 256, 0, stream>>>(input_seqs, inp16);
  k_transposeW<<<kD, 256, 0, stream>>>(emb2_w, emb2T, kD, kD);
  k_transposeW<<<kD, 256, 0, stream>>>(Wd_w,   WdT,   kD, kD);
  k_transposeW<<<kD, 256, 0, stream>>>(whk_w,  whkT,  kD, kD);
  k_buildWcat<<<kDG, 256, 0, stream>>>(Wall_w, Uall_w, WcatT);
  k_biascat<<<kDG / 256, 256, 0, stream>>>(Wall_b, Uall_b, biascat);

  // 2. x = input_seqs @ emb2 + b   (WMMA)
  k_gemm16<<<dim3(rows / 16, kD / 16), 32, 0, stream>>>(inp16, emb2T, emb2_b,
                                                        x32, x16, kD, kD);
  // 3. time feature
  k_timefeat<<<rows, 256, 0, stream>>>(seq_time, sel_w, sel_b, time_w, time_b, tf32);

  // 4. T-LSTM #1 (persistent WMMA scan)
  k_tlstm<<<kB / 16, 512, 0, stream>>>(x16, tf32, WcatT, biascat, WdT, Wd_b,
                                       outs32, outs16);
  // 5. whk = outputs @ whk_w + b (WMMA; row s=199 computed but unused)
  k_gemm16<<<dim3(rows / 16, kD / 16), 32, 0, stream>>>(outs16, whkT, whk_b,
                                                        whk32, (half_t*)nullptr, kD, kD);
  // 6. attention -> aligned
  k_copy_ek<<<kB, 256, 0, stream>>>(x32, align32);
  k_attn<<<dim3(kSeq - 1, kB), 64, 0, stream>>>(x32, whk32, w1_w, w1_b, w2_w, w2_b,
                                                align32);
  // 7. gen_x
  k_genx<<<(int)(kBSD / 256), 256, 0, stream>>>(align32, normal_n, pred_n, genx16);

  // 8. T-LSTM #2
  k_tlstm<<<kB / 16, 512, 0, stream>>>(genx16, tf32, WcatT, biascat, WdT, Wd_b,
                                       gouts32, (half_t*)nullptr);
  // 9. heads
  float* out_f = (float*)d_out;
  k_head<<<kB, 256, 0, stream>>>(outs32,  out_w, out_b, out_f);
  k_head<<<kB, 256, 0, stream>>>(gouts32, out_w, out_b, out_f + kB * 2);

  // 10. pass-through outputs (tuple order: out, gen_out, predicted_noise, normal_noise)
  (void)hipMemcpyAsync(out_f + 2 * kB * 2,        pred_n,   kBSD * 4,
                       hipMemcpyDeviceToDevice, stream);
  (void)hipMemcpyAsync(out_f + 2 * kB * 2 + kBSD, normal_n, kBSD * 4,
                       hipMemcpyDeviceToDevice, stream);
}
